// NeuralCAComplete_10883447128227
// MI455X (gfx1250) — matrix operand order repaired and checked by reading, not compiled, not run on hardware
//
#include <hip/hip_runtime.h>
#include <hip/hip_bf16.h>

// ---------------- problem constants (from reference) ----------------
#define CH     16      // state channels
#define HH     96      // height
#define WW     96      // width
#define KDIM   48      // 3*CH perception channels
#define HID    128     // hidden dim
#define TILE   16
#define HALO   18      // TILE + 2
#define TPW    6       // tiles per row/col = 96/16

typedef __attribute__((ext_vector_type(16))) _Float16 v16h;
typedef __attribute__((ext_vector_type(8)))  _Float16 v8h;
typedef __attribute__((ext_vector_type(8)))  float    v8f;

#if defined(__gfx1250__) && __has_builtin(__builtin_amdgcn_global_load_async_to_lds_b32)
#define USE_ASYNC_LDS 1
typedef __attribute__((address_space(1))) int gas_i32;   // global int
typedef __attribute__((address_space(3))) int las_i32;   // LDS int
#endif

// ---------------- threefry2x32 (JAX-style counter PRNG) ----------------
__device__ __forceinline__ unsigned tf_rotl(unsigned x, unsigned n) {
    return (x << n) | (x >> (32u - n));
}
struct U2 { unsigned x, y; };

__device__ __forceinline__ U2 threefry2x32(unsigned k0, unsigned k1,
                                           unsigned c0, unsigned c1) {
    const unsigned ks2 = k0 ^ k1 ^ 0x1BD11BDAu;
    unsigned x0 = c0 + k0, x1 = c1 + k1;
    const unsigned R0[4] = {13u, 15u, 26u, 6u};
    const unsigned R1[4] = {17u, 29u, 16u, 24u};
#pragma unroll
    for (int i = 0; i < 4; ++i) { x0 += x1; x1 = tf_rotl(x1, R0[i]); x1 ^= x0; }
    x0 += k1; x1 += ks2 + 1u;
#pragma unroll
    for (int i = 0; i < 4; ++i) { x0 += x1; x1 = tf_rotl(x1, R1[i]); x1 ^= x0; }
    x0 += ks2; x1 += k0 + 2u;
#pragma unroll
    for (int i = 0; i < 4; ++i) { x0 += x1; x1 = tf_rotl(x1, R0[i]); x1 ^= x0; }
    x0 += k0; x1 += k1 + 3u;
#pragma unroll
    for (int i = 0; i < 4; ++i) { x0 += x1; x1 = tf_rotl(x1, R1[i]); x1 ^= x0; }
    x0 += k1; x1 += ks2 + 4u;
#pragma unroll
    for (int i = 0; i < 4; ++i) { x0 += x1; x1 = tf_rotl(x1, R0[i]); x1 ^= x0; }
    x0 += ks2; x1 += k0 + 5u;
    return U2{x0, x1};
}

__device__ __forceinline__ float bits_to_unit(unsigned bits) {
    unsigned u = (bits >> 9) | 0x3F800000u;      // [1,2)
    return __uint_as_float(u) - 1.0f;            // [0,1)
}

// ---------------- perception: fixed depthwise 3x3 (id / sobel-x / sobel-y) ----
__device__ __forceinline__ float perceive(const float (*xs)[HALO][HALO],
                                          int k, int row, int col) {
    const int c = k / 3;
    const int d = k - 3 * c;
    if (d == 0) return xs[c][row + 1][col + 1];
    const float a00 = xs[c][row][col],     a01 = xs[c][row][col + 1],     a02 = xs[c][row][col + 2];
    const float a10 = xs[c][row + 1][col],                                a12 = xs[c][row + 1][col + 2];
    const float a20 = xs[c][row + 2][col], a21 = xs[c][row + 2][col + 1], a22 = xs[c][row + 2][col + 2];
    if (d == 1)  // sobel-x / 8 (cross-correlation, matches lax.conv)
        return ((a02 - a00) + 2.0f * (a12 - a10) + (a22 - a20)) * 0.125f;
    return ((a20 - a00) + 2.0f * (a21 - a01) + (a22 - a02)) * 0.125f;  // sobel-y / 8
}

// ---------------- weight fragment loaders (A-matrix 16x32 f16, contiguous-K order shared with B) ----
// Per lane: M = mtile*16 + (lane&15); halves 0..7 -> K = kb+h, halves 8..15 -> K = kb+8+h
// with kb = kchunk*32 + (lane>>4)*16.  Out-of-range K (>= kmax) are zero.
__device__ __forceinline__ v16h load_a_tile(const float* __restrict__ w,
                                            int rowstride, int kmax,
                                            int mbase, int kchunk, int ln, int g) {
    const float* rowp = w + (size_t)(mbase + ln) * rowstride;
    const int kb = kchunk * 32 + g * 16;
    v16h a;
#pragma unroll
    for (int h = 0; h < 8; ++h) {
        int k0 = kb + h;
        int k1 = kb + 8 + h;
        float f0 = rowp[k0 < kmax ? k0 : 0];
        float f1 = rowp[k1 < kmax ? k1 : 0];
        a[h]     = (k0 < kmax) ? (_Float16)f0 : (_Float16)0.0f;
        a[h + 8] = (k1 < kmax) ? (_Float16)f1 : (_Float16)0.0f;
    }
    return a;
}

// ---------------- kernel A: perception + WMMA GEMMs + stochastic update ------
__global__ __launch_bounds__(256) void ca_update_kernel(
        const float* __restrict__ xin,   // (B,16,96,96)
        const float* __restrict__ w2,    // (128,48)
        const float* __restrict__ w3,    // (16,128)
        float* __restrict__ xmid,        // (B,16,96,96) scratch
        float* __restrict__ alive_pre,   // (B,96,96)    scratch
        int step) {
    __shared__ float    xs[CH][HALO][HALO];   // x tile + halo-1, fp32 (20.7 KB)
    __shared__ _Float16 hs[8][TILE][HID];     // per-wave hidden pane     (32 KB)

    const int b   = blockIdx.y;
    const int tix = (blockIdx.x % TPW) * TILE;
    const int tiy = (blockIdx.x / TPW) * TILE;
    const int tid = threadIdx.x;

    // --- load halo tile (zero padding outside image) ---
    const float* xb = xin + (size_t)b * CH * HH * WW;
#ifdef USE_ASYNC_LDS
    // CDNA5 async memory->LDS path: no VGPR round trip, tracked on ASYNCcnt.
    for (int i = tid; i < CH * HALO * HALO; i += 256) {
        int c  = i / (HALO * HALO);
        int r  = (i / HALO) % HALO;
        int q  = i % HALO;
        int gy = tiy + r - 1, gx = tix + q - 1;
        if (gy >= 0 && gy < HH && gx >= 0 && gx < WW) {
            const float* src = xb + (size_t)c * HH * WW + gy * WW + gx;
            __builtin_amdgcn_global_load_async_to_lds_b32(
                (gas_i32*)src, (las_i32*)&xs[c][r][q], 0, 0);
        } else {
            xs[c][r][q] = 0.0f;          // zero-pad edge cells directly
        }
    }
#if __has_builtin(__builtin_amdgcn_s_wait_asynccnt)
    __builtin_amdgcn_s_wait_asynccnt(0);
#else
    asm volatile("s_wait_asynccnt 0x0" ::: "memory");
#endif
#else
    for (int i = tid; i < CH * HALO * HALO; i += 256) {
        int c  = i / (HALO * HALO);
        int r  = (i / HALO) % HALO;
        int q  = i % HALO;
        int gy = tiy + r - 1, gx = tix + q - 1;
        float v = 0.0f;
        if (gy >= 0 && gy < HH && gx >= 0 && gx < WW)
            v = xb[(size_t)c * HH * WW + gy * WW + gx];
        xs[c][r][q] = v;
    }
#endif
    __syncthreads();

    // --- alive_pre: maxpool3(old alpha) > thresh, one pixel per thread ---
    {
        int py = tid >> 4, px = tid & 15;
        float m = -1e30f;
#pragma unroll
        for (int dy = 0; dy < 3; ++dy)
#pragma unroll
            for (int dx = 0; dx < 3; ++dx)
                m = fmaxf(m, xs[3][py + dy][px + dx]);
        alive_pre[(size_t)b * HH * WW + (tiy + py) * WW + (tix + px)] =
            (m > 0.1f) ? 1.0f : 0.0f;
    }

    // per-step PRNG key (uniform across threads -> SALU)
    const U2 key = threefry2x32(0u, 42u, (unsigned)step, 0x9E3779B9u);

    const int wave = tid >> 5;
    const int lane = tid & 31;
    const int ln   = lane & 15;   // N (pixel col) / M row within fragment
    const int g    = lane >> 4;   // lane group

    for (int sub = 0; sub < 2; ++sub) {
        const int row = wave * 2 + sub;         // pixel row inside tile

        // --- build B fragments: y[k][pixel], K padded 48 -> 64 ---
        // half j of chunk kc holds K = kc*32 + g*16 + j at pixel (row, ln)
        v16h b0, b1;
#pragma unroll
        for (int j = 0; j < 16; ++j) {
            int k0 = g * 16 + j;                 // 0..31, always valid
            int k1 = 32 + g * 16 + j;            // 32..47 valid only for g==0
            b0[j] = (_Float16)perceive(xs, k0, row, ln);
            float y1 = perceive(xs, (k1 < KDIM) ? k1 : 0, row, ln);
            b1[j] = (k1 < KDIM) ? (_Float16)y1 : (_Float16)0.0f;
        }

        // --- GEMM1: h = relu(w2 @ y), M=128 (8 tiles), K=64 (2 chunks) ---
        v8f acc[8];
#pragma unroll
        for (int mo = 0; mo < 8; ++mo) {
            v8f c = {};
            v16h a0 = load_a_tile(w2, KDIM, KDIM, mo * 16, 0, ln, g);
            c = __builtin_amdgcn_wmma_f32_16x16x32_f16(false, a0, false, b0,
                                                       (short)0, c, false, false);
            v16h a1 = load_a_tile(w2, KDIM, KDIM, mo * 16, 1, ln, g);
            c = __builtin_amdgcn_wmma_f32_16x16x32_f16(false, a1, false, b1,
                                                       (short)0, c, false, false);
            acc[mo] = c;
        }

        // --- ReLU + C-layout -> B-layout transpose via per-wave LDS pane ---
        // lane holds D(M = r + 8g, N = ln); store hs[wave][n][k] row-major in K
#pragma unroll
        for (int mo = 0; mo < 8; ++mo) {
            v8h pk;
#pragma unroll
            for (int r = 0; r < 8; ++r)
                pk[r] = (_Float16)fmaxf(acc[mo][r], 0.0f);
            *(v8h*)&hs[wave][ln][mo * 16 + g * 8] = pk;   // one b128 store
        }
        asm volatile("s_wait_dscnt 0x0" ::: "memory");    // wave-local LDS RAW

        // --- GEMM2: out = w3 @ h, M=16, K=128 (4 chunks) ---
        v8f acc2 = {};
#pragma unroll
        for (int kc = 0; kc < 4; ++kc) {
            v16h a  = load_a_tile(w3, HID, HID, 0, kc, ln, g);
            v16h bb = *(const v16h*)&hs[wave][ln][kc * 32 + g * 16];
            acc2 = __builtin_amdgcn_wmma_f32_16x16x32_f16(false, a, false, bb,
                                                          (short)0, acc2, false, false);
        }

        // --- x_mid = x + (rand > 0.5) * update ---
        // One threefry call yields 2 lanes of randomness -> 4 calls / 8 channels.
        const int gy = tiy + row, gx = tix + ln;
        float* xmb = xmid + (size_t)b * CH * HH * WW + gy * WW + gx;
#pragma unroll
        for (int rp = 0; rp < 4; ++rp) {
            const int r0 = 2 * rp;
            const int c0 = r0 + 8 * g;                    // even channel of pair
            const unsigned flat0 =
                (unsigned)(((b * CH + c0) * HH + gy) * WW + gx);
            const U2 rnd = threefry2x32(key.x, key.y, flat0, 0x13371337u);
            const float u0 = bits_to_unit(rnd.x);
            const float u1 = bits_to_unit(rnd.y);
            const float x0 = xs[c0][row + 1][ln + 1];
            const float x1 = xs[c0 + 1][row + 1][ln + 1];
            xmb[(size_t)c0 * HH * WW] =
                x0 + ((u0 > 0.5f) ? acc2[r0] : 0.0f);
            xmb[(size_t)(c0 + 1) * HH * WW] =
                x1 + ((u1 > 0.5f) ? acc2[r0 + 1] : 0.0f);
        }
    }
}

// ---------------- kernel B: alive masking (needs global sync after A) --------
__global__ __launch_bounds__(256) void ca_mask_kernel(
        const float* __restrict__ xmid,
        const float* __restrict__ alive_pre,
        float* __restrict__ out, int batch) {
    const int idx = blockIdx.x * 256 + threadIdx.x;       // over B*H*W
    if (idx >= batch * HH * WW) return;
    const int b = idx / (HH * WW);
    const int p = idx % (HH * WW);
    const int y = p / WW, x = p % WW;

    const float* alpha = xmid + (size_t)b * CH * HH * WW + 3 * HH * WW;
    float m = -1e30f;
#pragma unroll
    for (int dy = -1; dy <= 1; ++dy) {
        int yy = y + dy;
        if (yy < 0 || yy >= HH) continue;
#pragma unroll
        for (int dx = -1; dx <= 1; ++dx) {
            int xx = x + dx;
            if (xx < 0 || xx >= WW) continue;
            m = fmaxf(m, alpha[yy * WW + xx]);
        }
    }
    const float keep = ((m > 0.1f) && (alive_pre[idx] > 0.5f)) ? 1.0f : 0.0f;
    const float* src = xmid + (size_t)b * CH * HH * WW + p;
    float*       dst = out  + (size_t)b * CH * HH * WW + p;
#pragma unroll
    for (int c = 0; c < CH; ++c)
        dst[(size_t)c * HH * WW] = src[(size_t)c * HH * WW] * keep;
}

// ---------------- host-side launch ----------------
extern "C" void kernel_launch(void* const* d_in, const int* in_sizes, int n_in,
                              void* d_out, int out_size, void* d_ws, size_t ws_size,
                              hipStream_t stream) {
    const float* x0 = (const float*)d_in[0];   // (B,16,96,96)
    // d_in[1] = num_steps (device scalar; T derived from out_size instead)
    // d_in[2] = dw_kernel (fixed sobel stack; baked into perceive())
    const float* w2 = (const float*)d_in[3];   // (128,48)
    const float* w3 = (const float*)d_in[4];   // (16,128)

    const int    B     = in_sizes[0] / (CH * HH * WW);
    const size_t frame = (size_t)B * CH * HH * WW;
    const int    T     = (int)((size_t)out_size / frame);

    float* xmid  = (float*)d_ws;               // frame floats
    float* alive = xmid + frame;               // B*H*W floats
    float* out   = (float*)d_out;

    const dim3 gridA(TPW * TPW, B);
    const int  gridB = (B * HH * WW + 255) / 256;

    for (int t = 0; t < T; ++t) {
        const float* xin = (t == 0) ? x0 : out + (size_t)(t - 1) * frame;
        ca_update_kernel<<<gridA, 256, 0, stream>>>(xin, w2, w3, xmid, alive, t);
        ca_mask_kernel<<<gridB, 256, 0, stream>>>(xmid, alive,
                                                  out + (size_t)t * frame, B);
    }
}
